// CEBlock_88983132439291
// MI455X (gfx1250) — compile-verified, matrix operand order checked by reading, not run on hardware
//
#include <hip/hip_runtime.h>
#include <hip/hip_bf16.h>

// ---------------- problem constants ----------------
#define BB      32
#define LENS_Tc 64
#define LENS_Sc 256
#define NTOK    576            // LENS_T + 2*LENS_S
#define DIMc    768
#define NHc     12
#define HDc     64
#define MLP_Hc  3072
#define KEEPc   180            // ceil(0.7*256)
#define REMc    76
#define N2c     424            // 64 + 2*180
#define SCALEc  0.125f         // 64^-0.5
#define EPSc    1e-5f

typedef __attribute__((ext_vector_type(16))) _Float16 v16h;
typedef __attribute__((ext_vector_type(8)))  _Float16 v8h;
typedef __attribute__((ext_vector_type(8)))  float    v8f;
typedef unsigned int v4u __attribute__((ext_vector_type(4)));
typedef int          v4i_t __attribute__((ext_vector_type(4)));
typedef int          v8i_t __attribute__((ext_vector_type(8)));
typedef _Float16 half_t;

// ---------------- TDM availability ----------------
#if defined(__has_builtin)
#  if __has_builtin(__builtin_amdgcn_tensor_load_to_lds) && \
      __has_builtin(__builtin_amdgcn_s_wait_tensorcnt)
#    define HAVE_TDM 1
#  else
#    define HAVE_TDM 0
#  endif
#else
#  define HAVE_TDM 0
#endif

// ---------------- WMMA helpers ----------------
// A-fragment (16x32 f16, MxK) per ISA 7.12.2: lane L holds row M=L&15,
// halves j<8 -> K = kb+j, j>=8 -> K = kb+16+(j-8), kb = k0 + 8*(L>=16).
// B-fragment (32x16, KxN) mirrors with N = L&15. Sources are row-major with
// K contiguous, so each fragment is two 16-byte loads.
__device__ __forceinline__ v16h load_frag_g(const half_t* __restrict__ base,
                                            int ld, int row0, int k0, int lane) {
  int r  = row0 + (lane & 15);
  int kb = k0 + ((lane >> 4) << 3);
  const half_t* p = base + (size_t)r * ld + kb;
  v8h lo = *(const v8h*)p;
  v8h hi = *(const v8h*)(p + 16);
  v16h f;
#pragma unroll
  for (int i = 0; i < 8; ++i) { f[i] = lo[i]; f[i + 8] = hi[i]; }
  return f;
}

// same fragment pattern, from a 64-wide row-major LDS tile
__device__ __forceinline__ v16h load_frag_lds(const half_t* base,
                                              int row0, int k0, int lane) {
  int r  = row0 + (lane & 15);
  int kb = k0 + ((lane >> 4) << 3);
  const half_t* p = base + r * 64 + kb;
  v8h lo = *(const v8h*)p;
  v8h hi = *(const v8h*)(p + 16);
  v16h f;
#pragma unroll
  for (int i = 0; i < 8; ++i) { f[i] = lo[i]; f[i + 8] = hi[i]; }
  return f;
}

__device__ __forceinline__ v8f wmma_f16(v16h a, v16h b, v8f c) {
  return __builtin_amdgcn_wmma_f32_16x16x32_f16(false, a, false, b,
                                                (short)0, c, false, false);
}

__device__ __forceinline__ float gelu_exact(float x) {
  return 0.5f * x * (1.f + erff(x * 0.70710678118654752f));
}

// low 32 bits of the shared-aperture address == LDS byte offset (ISA 10.2)
__device__ __forceinline__ unsigned int lds_byte_off(const void* p) {
  return (unsigned int)(size_t)p;
}

// ---------------- TDM: DMA a 64-row x 64-f16 tile (row stride K elems) to LDS ----
#if HAVE_TDM
__device__ __forceinline__ void tdm_load_tile_64x64(const half_t* gsrc, int K,
                                                    unsigned int lds_off) {
  unsigned long long ga = (unsigned long long)(size_t)gsrc;
  v4u g0;
  g0[0] = 1u;                                   // count=1, user descriptor
  g0[1] = lds_off;                              // LDS byte address
  g0[2] = (unsigned int)(ga & 0xffffffffu);     // global_addr[31:0]
  g0[3] = (unsigned int)((ga >> 32) & 0x01ffffffu) | (2u << 30);  // addr[56:32], type=2
  unsigned int td0 = (unsigned int)K;           // tensor_dim0 = row length (elems)
  unsigned int td1 = 1u << 20;                  // plenty of rows
  unsigned long long st0 = (unsigned long long)K;  // tensor_dim0_stride (elems)
  v8i_t g1;
  g1[0] = (int)(1u << 16);                      // data_size=1 (2B); mask/flags=0
  g1[1] = (int)((td0 & 0xffffu) << 16);         // [63:48] = tensor_dim0[15:0]
  g1[2] = (int)(((td0 >> 16) & 0xffffu) | ((td1 & 0xffffu) << 16));
  g1[3] = (int)(((td1 >> 16) & 0xffffu) | (64u << 16));   // tile_dim0 = 64
  g1[4] = (int)(64u);                           // tile_dim1 = 64, tile_dim2 = 0
  g1[5] = (int)(st0 & 0xffffffffu);
  g1[6] = (int)((st0 >> 32) & 0xffffu);         // dim1_stride = 0
  g1[7] = 0;
  v4i_t g2 = {0, 0, 0, 0};                      // dims 2/3 unused (tile_dim2=0)
  v4i_t g3 = {0, 0, 0, 0};
#if defined(__clang_major__) && (__clang_major__ >= 23)
  v8i_t g4 = {0, 0, 0, 0, 0, 0, 0, 0};
  __builtin_amdgcn_tensor_load_to_lds(g0, g1, g2, g3, g4, 0);
#else
  __builtin_amdgcn_tensor_load_to_lds(g0, g1, g2, g3, 0);
#endif
}
#endif

// fallback: cooperative 64x64 f16 tile copy (both tiles), vector 16B chunks
__device__ __forceinline__ void lds_stage_copy(const half_t* __restrict__ gA,
                                               const half_t* __restrict__ gB,
                                               int K, half_t* dA, half_t* dB, int tid) {
  for (int c = tid; c < 512; c += 256) {
    int r = c >> 3, o = (c & 7) * 8;
    *(v8h*)(dA + r * 64 + o) = *(const v8h*)(gA + (size_t)r * K + o);
    *(v8h*)(dB + r * 64 + o) = *(const v8h*)(gB + (size_t)r * K + o);
  }
}

// ---------------- double-buffered LDS/TDM-staged 64x64-tile GEMM core ----------
// block: 256 threads (8 waves), tile 64(M) x 64(N); wave does 16x32.
__device__ __forceinline__ void gemm_tile_staged(
    const half_t* __restrict__ A, const half_t* __restrict__ W, int K,
    int Mblk, int Nblk, int wave, int lane, int tid,
    half_t* lsA, half_t* lsB, v8f& acc0, v8f& acc1) {
  int Mloc = (wave & 3) * 16, Nloc = (wave >> 2) * 32;
  int nst = K >> 6;
#if HAVE_TDM
  if (wave == 0) {
    tdm_load_tile_64x64(A + (size_t)Mblk * K, K, lds_byte_off(lsA));
    tdm_load_tile_64x64(W + (size_t)Nblk * K, K, lds_byte_off(lsB));
  }
#else
  lds_stage_copy(A + (size_t)Mblk * K, W + (size_t)Nblk * K, K, lsA, lsB, tid);
#endif
  for (int kc = 0; kc < nst; ++kc) {
    int buf = kc & 1;
    bool more = (kc + 1) < nst;
#if HAVE_TDM
    if (more && wave == 0) {
      int ko = (kc + 1) << 6;
      tdm_load_tile_64x64(A + (size_t)Mblk * K + ko, K,
                          lds_byte_off(lsA) + (unsigned)(buf ^ 1) * 8192u);
      tdm_load_tile_64x64(W + (size_t)Nblk * K + ko, K,
                          lds_byte_off(lsB) + (unsigned)(buf ^ 1) * 8192u);
    }
    if (wave == 0) {
      if (more) __builtin_amdgcn_s_wait_tensorcnt(2);   // stage kc done (kc+1 in flight)
      else      __builtin_amdgcn_s_wait_tensorcnt(0);   // final stage done
    }
#endif
    __syncthreads();                       // stage kc visible to all waves
    half_t* tA = lsA + buf * 4096;
    half_t* tB = lsB + buf * 4096;
#pragma unroll
    for (int ks = 0; ks < 64; ks += 32) {
      v16h a  = load_frag_lds(tA, Mloc, ks, lane);
      v16h b0 = load_frag_lds(tB, Nloc, ks, lane);
      v16h b1 = load_frag_lds(tB, Nloc + 16, ks, lane);
      acc0 = wmma_f16(a, b0, acc0);
      acc1 = wmma_f16(a, b1, acc1);
    }
    __syncthreads();                       // done reading buf before it is refilled
#if !HAVE_TDM
    if (more) {
      int ko = (kc + 1) << 6;
      lds_stage_copy(A + (size_t)Mblk * K + ko, W + (size_t)Nblk * K + ko, K,
                     lsA + (buf ^ 1) * 4096, lsB + (buf ^ 1) * 4096, tid);
    }
#endif
  }
}

// ---------------- small utility kernels ----------------
__global__ void f32_to_f16_kernel(const float* __restrict__ in,
                                  half_t* __restrict__ out, long n) {
  long i = (long)blockIdx.x * blockDim.x + threadIdx.x;
  long stride = (long)gridDim.x * blockDim.x;
  for (; i < n; i += stride) out[i] = (half_t)in[i];
}

// one block per row of DIM floats -> f16 normalized row
__global__ void layernorm_f16_kernel(const float* __restrict__ x,
                                     const float* __restrict__ sc,
                                     const float* __restrict__ sh,
                                     half_t* __restrict__ out) {
  __shared__ float red[256];
  int row = blockIdx.x;
  const float* xr = x + (size_t)row * DIMc;
  float sum = 0.f;
  for (int c = threadIdx.x; c < DIMc; c += 256) sum += xr[c];
  red[threadIdx.x] = sum; __syncthreads();
  for (int o = 128; o > 0; o >>= 1) {
    if (threadIdx.x < o) red[threadIdx.x] += red[threadIdx.x + o];
    __syncthreads();
  }
  float mean = red[0] * (1.f / DIMc);
  __syncthreads();
  float vs = 0.f;
  for (int c = threadIdx.x; c < DIMc; c += 256) { float d = xr[c] - mean; vs += d * d; }
  red[threadIdx.x] = vs; __syncthreads();
  for (int o = 128; o > 0; o >>= 1) {
    if (threadIdx.x < o) red[threadIdx.x] += red[threadIdx.x + o];
    __syncthreads();
  }
  float rstd = rsqrtf(red[0] * (1.f / DIMc) + EPSc);
  half_t* orow = out + (size_t)row * DIMc;
  for (int c = threadIdx.x; c < DIMc; c += 256)
    orow[c] = (half_t)((xr[c] - mean) * rstd * sc[c] + sh[c]);
}

// ---------------- QKV GEMM (M=B*N, N=2304, K=768), scatter epilogue ----------------
__global__ void qkv_gemm_kernel(const half_t* __restrict__ A,
                                const half_t* __restrict__ W,
                                half_t* __restrict__ q16, half_t* __restrict__ k16,
                                half_t* __restrict__ vT16,
                                float* __restrict__ k_out, float* __restrict__ v_out) {
  __shared__ half_t lsA[2 * 4096];
  __shared__ half_t lsB[2 * 4096];
  int tid = threadIdx.x, lane = tid & 31, wave = tid >> 5;
  int Mblk = blockIdx.x * 64, Nblk = blockIdx.y * 64;
  v8f acc0 = {}, acc1 = {};
  gemm_tile_staged(A, W, DIMc, Mblk, Nblk, wave, lane, tid, lsA, lsB, acc0, acc1);
  int Mloc = (wave & 3) * 16, Nloc = (wave >> 2) * 32;
  int mlo = (lane >> 4) << 3, nc = lane & 15;
#pragma unroll
  for (int i = 0; i < 8; ++i) {
    int gr = Mblk + Mloc + mlo + i;
    int b = gr / NTOK, n = gr % NTOK;
#pragma unroll
    for (int t2 = 0; t2 < 2; ++t2) {
      int gc = Nblk + Nloc + t2 * 16 + nc;
      float val = t2 ? acc1[i] : acc0[i];
      int t = gc / DIMc, rem = gc % DIMc;
      int hh = rem >> 6, d = rem & 63;
      size_t bhnd = (((size_t)(b * NHc + hh)) * NTOK + n) * HDc + d;
      if (t == 0)       { q16[bhnd] = (half_t)val; }
      else if (t == 1)  { k16[bhnd] = (half_t)val; k_out[bhnd] = val; }
      else              { vT16[(((size_t)(b * NHc + hh)) * HDc + d) * NTOK + n] = (half_t)val;
                          v_out[bhnd] = val; }
    }
  }
}

// ---------------- fused attention: scores -> softmax -> attn out -> PV ----------------
// grid: (B*NH, 36) ; block 256 (8 waves); one block = 16 queries vs all 576 keys
__global__ void attn_kernel(const half_t* __restrict__ q16,
                            const half_t* __restrict__ k16,
                            const half_t* __restrict__ vT16,
                            float* __restrict__ attn_out,
                            half_t* __restrict__ xa16) {
  __shared__ float s[16][NTOK];     // 36 KB probability stripe
  __shared__ float red[16][16];
  __shared__ float rowv[16];
  int bh = blockIdx.x;              // b*NH + h
  int qt = blockIdx.y;              // query tile 0..35
  int lane = threadIdx.x & 31, wave = threadIdx.x >> 5;
  const half_t* qb = q16 + (size_t)bh * NTOK * HDc;
  const half_t* kb = k16 + (size_t)bh * NTOK * HDc;
  int q0 = qt * 16;

  // ---- phase A: S = (Q K^T) * scale, via WMMA (K = HD = 64 -> 2 wmma/tile)
  for (int kt = wave; kt < 36; kt += 8) {
    v8f acc = {};
    for (int k0 = 0; k0 < HDc; k0 += 32) {
      v16h a = load_frag_g(qb, HDc, q0, k0, lane);
      v16h b = load_frag_g(kb, HDc, kt * 16, k0, lane);
      acc = wmma_f16(a, b, acc);
    }
    int mlo = (lane >> 4) << 3, nc = lane & 15;
#pragma unroll
    for (int i = 0; i < 8; ++i) s[mlo + i][kt * 16 + nc] = acc[i] * SCALEc;
  }
  __syncthreads();

  // ---- phase B: softmax over 576 per row (16 rows x 16 chunks, 36 elems each)
  int row = threadIdx.x >> 4, chunk = threadIdx.x & 15;
  float m = -3.4e38f;
  for (int i = 0; i < 36; ++i) m = fmaxf(m, s[row][chunk + (i << 4)]);
  red[row][chunk] = m; __syncthreads();
  if (threadIdx.x < 16) {
    float mm = red[threadIdx.x][0];
    for (int j = 1; j < 16; ++j) mm = fmaxf(mm, red[threadIdx.x][j]);
    rowv[threadIdx.x] = mm;
  }
  __syncthreads();
  float rmax = rowv[row];
  float psum = 0.f;
  for (int i = 0; i < 36; ++i) {
    int c = chunk + (i << 4);
    float e = __expf(s[row][c] - rmax);
    s[row][c] = e; psum += e;
  }
  __syncthreads();
  red[row][chunk] = psum; __syncthreads();
  if (threadIdx.x < 16) {
    float ss = 0.f;
    for (int j = 0; j < 16; ++j) ss += red[threadIdx.x][j];
    rowv[threadIdx.x] = 1.f / ss;
  }
  __syncthreads();
  float rinv = rowv[row];
  float* arow = attn_out + ((size_t)bh * NTOK + (q0 + row)) * NTOK;
  for (int i = 0; i < 36; ++i) {
    int c = chunk + (i << 4);
    float p = s[row][c] * rinv;
    s[row][c] = p;
    arow[c] = p;                 // 509 MB streaming store (required output)
  }
  __syncthreads();

  // ---- phase C: xa = P @ V (K = 576 -> 18 wmma), waves 0..3 cover HD=64
  if (wave < 4) {
    int d0 = wave * 16;
    const half_t* vb = vT16 + (size_t)bh * HDc * NTOK;   // [d][key], key contiguous
    v8f acc = {};
    for (int k0 = 0; k0 < NTOK; k0 += 32) {
      int r = lane & 15;
      int kbo = k0 + ((lane >> 4) << 3);
      v16h a;
#pragma unroll
      for (int i = 0; i < 8; ++i) {
        a[i]     = (half_t)s[r][kbo + i];
        a[i + 8] = (half_t)s[r][kbo + 16 + i];
      }
      v16h b = load_frag_g(vb, NTOK, d0, k0, lane);
      acc = wmma_f16(a, b, acc);
    }
    int mlo = (lane >> 4) << 3, nc = lane & 15;
    int b_  = bh / NHc, h_ = bh % NHc;
#pragma unroll
    for (int i = 0; i < 8; ++i) {
      int qrow = q0 + mlo + i;
      xa16[((size_t)(b_ * NTOK + qrow)) * DIMc + h_ * HDc + d0 + nc] = (half_t)acc[i];
    }
  }
}

// ---------------- GEMM + bias + residual -> fp32 (proj, fc2) ----------------
__global__ void gemm_bias_resid_kernel(const half_t* __restrict__ A,
                                       const half_t* __restrict__ W,
                                       const float* __restrict__ bias,
                                       const float* __restrict__ resid,
                                       float* __restrict__ out, int K, int Nld) {
  __shared__ half_t lsA[2 * 4096];
  __shared__ half_t lsB[2 * 4096];
  int tid = threadIdx.x, lane = tid & 31, wave = tid >> 5;
  int Mblk = blockIdx.x * 64, Nblk = blockIdx.y * 64;
  v8f acc0 = {}, acc1 = {};
  gemm_tile_staged(A, W, K, Mblk, Nblk, wave, lane, tid, lsA, lsB, acc0, acc1);
  int Mloc = (wave & 3) * 16, Nloc = (wave >> 2) * 32;
  int mlo = (lane >> 4) << 3, nc = lane & 15;
#pragma unroll
  for (int i = 0; i < 8; ++i) {
    int gr = Mblk + Mloc + mlo + i;
    int gc0 = Nblk + Nloc + nc, gc1 = Nblk + Nloc + 16 + nc;
    out[(size_t)gr * Nld + gc0] = resid[(size_t)gr * Nld + gc0] + bias[gc0] + acc0[i];
    out[(size_t)gr * Nld + gc1] = resid[(size_t)gr * Nld + gc1] + bias[gc1] + acc1[i];
  }
}

// ---------------- GEMM + bias + GELU -> f16 (fc1) ----------------
__global__ void gemm_bias_gelu_f16_kernel(const half_t* __restrict__ A,
                                          const half_t* __restrict__ W,
                                          const float* __restrict__ bias,
                                          half_t* __restrict__ out, int K, int Nld) {
  __shared__ half_t lsA[2 * 4096];
  __shared__ half_t lsB[2 * 4096];
  int tid = threadIdx.x, lane = tid & 31, wave = tid >> 5;
  int Mblk = blockIdx.x * 64, Nblk = blockIdx.y * 64;
  v8f acc0 = {}, acc1 = {};
  gemm_tile_staged(A, W, K, Mblk, Nblk, wave, lane, tid, lsA, lsB, acc0, acc1);
  int Mloc = (wave & 3) * 16, Nloc = (wave >> 2) * 32;
  int mlo = (lane >> 4) << 3, nc = lane & 15;
#pragma unroll
  for (int i = 0; i < 8; ++i) {
    int gr = Mblk + Mloc + mlo + i;
    int gc0 = Nblk + Nloc + nc, gc1 = Nblk + Nloc + 16 + nc;
    out[(size_t)gr * Nld + gc0] = (half_t)gelu_exact(acc0[i] + bias[gc0]);
    out[(size_t)gr * Nld + gc1] = (half_t)gelu_exact(acc1[i] + bias[gc1]);
  }
}

// ---------------- attn_t = mean over heads & first 64 queries ----------------
__global__ void attn_mean_kernel(const float* __restrict__ attn,
                                 float* __restrict__ scores) {
  int idx = blockIdx.x * blockDim.x + threadIdx.x;     // b*512 + j
  if (idx >= BB * 2 * LENS_Sc) return;
  int b = idx >> 9, j = idx & 511;
  float sum = 0.f;
  for (int h = 0; h < NHc; ++h) {
    const float* base = attn + (size_t)(b * NHc + h) * NTOK * NTOK + (LENS_Tc + j);
    for (int q = 0; q < LENS_Tc; ++q) sum += base[(size_t)q * NTOK];
  }
  scores[idx] = sum * (1.f / (NHc * LENS_Tc));
}

// ---------------- stable argsort(-scores) + gather (one block per (b,group)) ----
__global__ void prune_kernel(const float* __restrict__ scores,
                             const int* __restrict__ gidx_ps,
                             const int* __restrict__ gidx_s,
                             const float* __restrict__ x2,
                             float* __restrict__ x3,
                             int* __restrict__ keep_ps, int* __restrict__ keep_s,
                             int* __restrict__ rem_ps,  int* __restrict__ rem_s) {
  __shared__ float sc[256];
  __shared__ int order[256];
  int b = blockIdx.x >> 1, g = blockIdx.x & 1;
  int tid = threadIdx.x;
  sc[tid] = scores[b * 512 + g * 256 + tid];
  __syncthreads();
  float si = sc[tid];
  int rank = 0;
  for (int j = 0; j < 256; ++j) {
    float sj = sc[j];
    rank += (sj > si) || (sj == si && j < tid);   // stable descending
  }
  order[rank] = tid;
  __syncthreads();
  const int* gidx = (g == 0 ? gidx_ps : gidx_s) + b * LENS_Sc;
  int src = order[tid];
  if (tid < KEEPc) (g == 0 ? keep_ps : keep_s)[b * KEEPc + tid] = gidx[src];
  else             (g == 0 ? rem_ps  : rem_s )[b * REMc + (tid - KEEPc)] = gidx[src];
  // gather kept token rows
  const float* srcb = x2 + ((size_t)b * NTOK + LENS_Tc + g * LENS_Sc) * DIMc;
  float* dstb = x3 + ((size_t)b * N2c + LENS_Tc + g * KEEPc) * DIMc;
  for (int r = 0; r < KEEPc; ++r) {
    int sr = order[r];
    const float* srow = srcb + (size_t)sr * DIMc;
    float* drow = dstb + (size_t)r * DIMc;
    for (int c = tid; c < DIMc; c += 256) drow[c] = srow[c];
  }
}

// ---------------- copy t-tokens + gidx_t passthrough ----------------
__global__ void copy_t_kernel(const float* __restrict__ x2, float* __restrict__ x3,
                              const int* __restrict__ gidx_t, int* __restrict__ gidx_t_out) {
  long i = (long)blockIdx.x * 256 + threadIdx.x;
  long nt = (long)BB * LENS_Tc * DIMc;
  long stride = (long)gridDim.x * 256;
  for (long p = i; p < nt; p += stride) {
    long b = p / (LENS_Tc * DIMc);
    long rc = p % (LENS_Tc * DIMc);
    x3[(size_t)b * N2c * DIMc + rc] = x2[(size_t)b * NTOK * DIMc + rc];
  }
  if (i < BB * LENS_Tc) gidx_t_out[i] = gidx_t[i];
}

// =========================== launcher ===========================
extern "C" void kernel_launch(void* const* d_in, const int* in_sizes, int n_in,
                              void* d_out, int out_size, void* d_ws, size_t ws_size,
                              hipStream_t stream) {
  (void)in_sizes; (void)n_in; (void)out_size; (void)ws_size;
  const float* x      = (const float*)d_in[0];
  const int*   gidx_t = (const int*)  d_in[1];
  const int*   gidx_ps= (const int*)  d_in[2];
  const int*   gidx_s = (const int*)  d_in[3];
  const float* qkv_w  = (const float*)d_in[4];
  const float* proj_w = (const float*)d_in[5];
  const float* proj_b = (const float*)d_in[6];
  const float* n1_s   = (const float*)d_in[7];
  const float* n1_b   = (const float*)d_in[8];
  const float* n2_s   = (const float*)d_in[9];
  const float* n2_b   = (const float*)d_in[10];
  const float* fc1_w  = (const float*)d_in[11];
  const float* fc1_b  = (const float*)d_in[12];
  const float* fc2_w  = (const float*)d_in[13];
  const float* fc2_b  = (const float*)d_in[14];

  const long M1 = (long)BB * NTOK;    // 18432
  const long M2 = (long)BB * N2c;     // 13568

  // output layout (tuple order, flat)
  float* out_x      = (float*)d_out;                          // 32*424*768
  int*   out_gidx_t = (int*)(out_x + (size_t)M2 * DIMc);      // 2048
  int*   out_keep_ps= out_gidx_t + BB * LENS_Tc;              // 5760
  int*   out_keep_s = out_keep_ps + BB * KEEPc;               // 5760
  int*   out_rem_ps = out_keep_s + BB * KEEPc;                // 2432
  int*   out_rem_s  = out_rem_ps + BB * REMc;                 // 2432
  float* out_attn   = (float*)(out_rem_s + BB * REMc);        // 32*12*576*576
  float* out_k      = out_attn + (size_t)BB * NHc * NTOK * NTOK;
  float* out_v      = out_k + (size_t)BB * NHc * NTOK * HDc;

  // workspace carve (256B aligned)
  char* wp = (char*)d_ws;
  auto carve = [&](size_t bytes) -> void* {
    void* p = (void*)wp; wp += (bytes + 255) & ~(size_t)255; return p;
  };
  half_t* h16    = (half_t*)carve((size_t)M1 * DIMc * 2);
  half_t* qkvw16 = (half_t*)carve((size_t)3 * DIMc * DIMc * 2);
  half_t* projw16= (half_t*)carve((size_t)DIMc * DIMc * 2);
  half_t* fc1w16 = (half_t*)carve((size_t)MLP_Hc * DIMc * 2);
  half_t* fc2w16 = (half_t*)carve((size_t)DIMc * MLP_Hc * 2);
  half_t* q16    = (half_t*)carve((size_t)BB * NHc * NTOK * HDc * 2);
  half_t* k16    = (half_t*)carve((size_t)BB * NHc * NTOK * HDc * 2);
  half_t* vT16   = (half_t*)carve((size_t)BB * NHc * HDc * NTOK * 2);
  half_t* xa16   = (half_t*)carve((size_t)M1 * DIMc * 2);
  float*  x2     = (float*) carve((size_t)M1 * DIMc * 4);
  float*  sc_ws  = (float*) carve((size_t)BB * 512 * 4);
  float*  x3     = (float*) carve((size_t)M2 * DIMc * 4);
  half_t* h2_16  = (half_t*)carve((size_t)M2 * DIMc * 2);
  half_t* hh16   = (half_t*)carve((size_t)M2 * MLP_Hc * 2);

  // 1) f16 shadows of weights
  f32_to_f16_kernel<<<1024, 256, 0, stream>>>(qkv_w,  qkvw16, (long)3 * DIMc * DIMc);
  f32_to_f16_kernel<<<1024, 256, 0, stream>>>(proj_w, projw16,(long)DIMc * DIMc);
  f32_to_f16_kernel<<<1024, 256, 0, stream>>>(fc1_w,  fc1w16, (long)MLP_Hc * DIMc);
  f32_to_f16_kernel<<<1024, 256, 0, stream>>>(fc2_w,  fc2w16, (long)DIMc * MLP_Hc);

  // 2) LN1 -> f16
  layernorm_f16_kernel<<<(int)M1, 256, 0, stream>>>(x, n1_s, n1_b, h16);

  // 3) QKV GEMM (TDM-staged; scatter q/k/vT f16, k/v fp32 outputs)
  qkv_gemm_kernel<<<dim3(M1 / 64, (3 * DIMc) / 64), 256, 0, stream>>>(
      h16, qkvw16, q16, k16, vT16, out_k, out_v);

  // 4) fused attention (scores+softmax+attn out+PV)
  attn_kernel<<<dim3(BB * NHc, NTOK / 16), 256, 0, stream>>>(
      q16, k16, vT16, out_attn, xa16);

  // 5) proj + residual -> x2
  gemm_bias_resid_kernel<<<dim3(M1 / 64, DIMc / 64), 256, 0, stream>>>(
      xa16, projw16, proj_b, x, x2, DIMc, DIMc);

  // 6) pruning scores from attn
  attn_mean_kernel<<<(BB * 2 * LENS_Sc) / 256, 256, 0, stream>>>(out_attn, sc_ws);

  // 7) stable top-k prune + gather (per (b,group))
  prune_kernel<<<BB * 2, 256, 0, stream>>>(sc_ws, gidx_ps, gidx_s, x2, x3,
                                           out_keep_ps, out_keep_s,
                                           out_rem_ps, out_rem_s);

  // 8) t-token copy + gidx_t passthrough
  copy_t_kernel<<<2048, 256, 0, stream>>>(x2, x3, gidx_t, out_gidx_t);

  // 9) LN2 -> f16
  layernorm_f16_kernel<<<(int)M2, 256, 0, stream>>>(x3, n2_s, n2_b, h2_16);

  // 10) fc1 + GELU -> f16
  gemm_bias_gelu_f16_kernel<<<dim3(M2 / 64, MLP_Hc / 64), 256, 0, stream>>>(
      h2_16, fc1w16, fc1_b, hh16, DIMc, MLP_Hc);

  // 11) fc2 + bias + residual -> final x output
  gemm_bias_resid_kernel<<<dim3(M2 / 64, DIMc / 64), 256, 0, stream>>>(
      hh16, fc2w16, fc2_b, x3, out_x, MLP_Hc, DIMc);
}
